// QuantizedLinear_90391881712146
// MI455X (gfx1250) — matrix operand order repaired and checked by reading, not compiled, not run on hardware
//
#include <hip/hip_runtime.h>
#include <stdint.h>

// GEMM: out[M,N] = x[M,K] @ W[N,K]^T + bias,  W = dequant(int4 packed, per-128-group scale)
#define IN_F   4096
#define OUT_F  4096
#define GROUP  128
#define BM     128
#define BN     128
#define BK     32
#define LDA    40   // halves per LDS row (32 + 8 pad -> 80B stride, bank-conflict friendly)
#define LDB    40

typedef _Float16 half2v __attribute__((ext_vector_type(2)));
typedef _Float16 half8  __attribute__((ext_vector_type(8)));
typedef _Float16 v16h   __attribute__((ext_vector_type(16)));
typedef float    v8f    __attribute__((ext_vector_type(8)));
typedef float    f32x4  __attribute__((ext_vector_type(4)));

// NOTE on K ordering: both A and B tiles are stored in LDS with the per-8-group
// K permutation [0,4,1,5,2,6,3,7]. The GEMM reduces over K, so any permutation
// applied identically to A and B is correct — this one makes the int4 unpack
// (nibble pairs (j, j+4) from each packed dword) land in contiguous f16 pairs
// with zero shuffle instructions.

struct Stage {
  f32x4    a[4];    // 16 fp32 activations (one half-row of the 128x32 A tile)
  uint32_t b0, b1;  // 16 packed int4 weights
  float    scale;   // group scale for this (row, ktile)
};

__device__ __forceinline__ v16h cat16(half8 lo, half8 hi) {
  return __builtin_shufflevector(lo, hi, 0,1,2,3,4,5,6,7,8,9,10,11,12,13,14,15);
}

// Magic-number dequant: ((w>>s) & 0x000F000F) ^ 0x64086408 is the f16 pair
// (1024 + (v^8)) for the two nibbles; signed int4 value = (v^8) - 8, so
// result = pair * scale + (-1032*scale), one v_pk_fma_f16 per 2 elements.
__device__ __forceinline__ half8 dq8(uint32_t w, half2v sc2, half2v c2) {
  half8 r;
#pragma unroll
  for (int j = 0; j < 4; ++j) {
    uint32_t t = ((w >> (4 * j)) & 0x000F000Fu) ^ 0x64086408u;
    half2v h = __builtin_bit_cast(half2v, t);
    half2v d = h * sc2 + c2;                 // v_pk_fma_f16
    r[2 * j]     = d[0];
    r[2 * j + 1] = d[1];
  }
  return r;
}

__device__ __forceinline__ void issue_loads(Stage& s,
    const float* __restrict__ x, const uint8_t* __restrict__ qw,
    const float* __restrict__ scales, int mBase, int nBase, int k, int tid)
{
  const int row = tid >> 1;   // 0..127
  const int h   = tid & 1;    // which 16-wide K half of the 32-wide tile
  const float* ap = x + (size_t)(mBase + row) * IN_F + k + h * 16;
  s.a[0] = *(const f32x4*)(ap + 0);
  s.a[1] = *(const f32x4*)(ap + 4);
  s.a[2] = *(const f32x4*)(ap + 8);
  s.a[3] = *(const f32x4*)(ap + 12);
  const int o = nBase + row;  // output-feature (W row)
  const uint32_t* bp = (const uint32_t*)(qw + (size_t)o * (IN_F / 2) + (k >> 1) + h * 8);
  s.b0 = bp[0];
  s.b1 = bp[1];
  s.scale = scales[o * (IN_F / GROUP) + (k >> 7)];  // BK=32 sits inside one 128-group
}

__device__ __forceinline__ void store_tile(const Stage& s,
    _Float16* __restrict__ Al, _Float16* __restrict__ Bl, int tid)
{
  const int row = tid >> 1;
  const int h   = tid & 1;

  // A: fp32 -> f16 with the [0,4,1,5,2,6,3,7] per-group K permutation baked in
  // (just a different pairing of floats into v_cvt_pk_f16_f32 — free).
  _Float16* ad = Al + row * LDA + h * 16;
  half8 a0, a1;
#pragma unroll
  for (int i = 0; i < 4; ++i) {
    a0[2 * i]     = (_Float16)s.a[0][i];   // true k-offset i      -> pos 2i
    a0[2 * i + 1] = (_Float16)s.a[1][i];   // true k-offset i + 4  -> pos 2i+1
    a1[2 * i]     = (_Float16)s.a[2][i];   // true k-offset i + 8
    a1[2 * i + 1] = (_Float16)s.a[3][i];   // true k-offset i + 12
  }
  *(half8*)(ad + 0) = a0;
  *(half8*)(ad + 8) = a1;

  // B: int4 symmetric dequant -> f16 via magic-number pk_fma path
  const float scf = s.scale;
  const _Float16 sch = (_Float16)scf;
  const _Float16 cch = (_Float16)(scf * -1032.0f);
  half2v sc2; sc2[0] = sch; sc2[1] = sch;
  half2v c2;  c2[0]  = cch; c2[1]  = cch;

  _Float16* bd = Bl + row * LDB + h * 16;
  *(half8*)(bd + 0) = dq8(s.b0, sc2, c2);
  *(half8*)(bd + 8) = dq8(s.b1, sc2, c2);
}

__global__ __launch_bounds__(256)
void qlinear_int4_wmma(const float* __restrict__ x,
                       const uint8_t* __restrict__ qw,
                       const float* __restrict__ scales,
                       const float* __restrict__ bias,
                       float* __restrict__ out)
{
  __shared__ _Float16 Al[2][BM * LDA];   // double-buffered activation tile (f16)
  __shared__ _Float16 Bl[2][BN * LDB];   // double-buffered dequantized weight tile (f16)

  const int tid   = threadIdx.x;
  const int mBase = blockIdx.y * BM;
  const int nBase = blockIdx.x * BN;

  const int lane = tid & 31;
  const int wv   = tid >> 5;        // wave 0..7
  const int wm   = (wv & 1) * 64;   // wave M offset in block tile
  const int wn   = (wv >> 1) * 32;  // wave N offset in block tile
  const int l16  = lane & 15;
  const int hseg = lane >> 4;       // 0 = lanes 0-15, 1 = lanes 16-31

  v8f acc[4][2] = {};               // 4 (M) x 2 (N) C fragments per wave

  Stage st;
  issue_loads(st, x, qw, scales, mBase, nBase, 0, tid);
  store_tile(st, Al[0], Bl[0], tid);

  const int NT = IN_F / BK;         // 128 K-steps
#pragma unroll 1
  for (int kt = 0; kt < NT; ++kt) {
    __syncthreads();
    const int  cur     = kt & 1;
    const bool hasNext = (kt + 1) < NT;

    Stage nx;
    if (hasNext) issue_loads(nx, x, qw, scales, mBase, nBase, (kt + 1) * BK, tid);

    const _Float16* Ac = Al[cur];
    const _Float16* Bc = Bl[cur];

    // B fragments: lane = column n (lane%16), K range = hseg*16 .. +15 (contiguous)
    v16h bfrag[2];
#pragma unroll
    for (int nb = 0; nb < 2; ++nb) {
      const _Float16* p = Bc + (wn + nb * 16 + l16) * LDB + hseg * 16;
      bfrag[nb] = cat16(*(const half8*)p, *(const half8*)(p + 8));
    }

    // A fragments: lane = row m (lane%16), K range = hseg*16 .. +15 (contiguous, matches B)
#pragma unroll
    for (int mr = 0; mr < 4; ++mr) {
      const _Float16* p = Ac + (wm + mr * 16 + l16) * LDA + hseg * 16;
      v16h afrag = cat16(*(const half8*)p, *(const half8*)(p + 8));
      acc[mr][0] = __builtin_amdgcn_wmma_f32_16x16x32_f16(
          false, afrag, false, bfrag[0], (short)0, acc[mr][0], false, false);
      acc[mr][1] = __builtin_amdgcn_wmma_f32_16x16x32_f16(
          false, afrag, false, bfrag[1], (short)0, acc[mr][1], false, false);
    }

    if (hasNext) store_tile(nx, Al[cur ^ 1], Bl[cur ^ 1], tid);
  }

  // Epilogue: C/D layout is VGPR v -> row (v + 8*hseg), lane%16 -> column. Add bias, store fp32.
#pragma unroll
  for (int nb = 0; nb < 2; ++nb) {
    const int   col = nBase + wn + nb * 16 + l16;
    const float bv  = bias[col];
#pragma unroll
    for (int mr = 0; mr < 4; ++mr) {
#pragma unroll
      for (int v = 0; v < 8; ++v) {
        const int rowM = mBase + wm + mr * 16 + hseg * 8 + v;
        out[(size_t)rowM * OUT_F + col] = acc[mr][nb][v] + bv;
      }
    }
  }
}

extern "C" void kernel_launch(void* const* d_in, const int* in_sizes, int n_in,
                              void* d_out, int out_size, void* d_ws, size_t ws_size,
                              hipStream_t stream)
{
  const float*   x      = (const float*)d_in[0];
  const uint8_t* qw     = (const uint8_t*)d_in[1];   // uint8 nibble-packed int4
  const float*   scales = (const float*)d_in[2];
  const float*   bias   = (const float*)d_in[3];
  float*         out    = (float*)d_out;

  const int M = in_sizes[0] / IN_F;     // 8*2048 = 16384
  dim3 grid(OUT_F / BN, M / BM);        // (32, 128)
  dim3 block(256);                      // 8 wave32 waves
  qlinear_int4_wmma<<<grid, block, 0, stream>>>(x, qw, scales, bias, out);
  (void)n_in; (void)out_size; (void)d_ws; (void)ws_size;
}
